// PQN_51299089383967
// MI455X (gfx1250) — matrix operand order repaired and checked by reading, not compiled, not run on hardware
//
#include <hip/hip_runtime.h>
#include <hip/hip_bf16.h>

typedef __attribute__((ext_vector_type(16))) _Float16 v16h;
typedef __attribute__((ext_vector_type(8)))  _Float16 v8h;
typedef __attribute__((ext_vector_type(8)))  float    v8f;
typedef __attribute__((ext_vector_type(4)))  unsigned int u32x4;
typedef __attribute__((ext_vector_type(8)))  int          i32x8;
typedef __attribute__((ext_vector_type(4)))  int          i32x4;

#define PQN_D     1024
#define PQN_L     128
#define PQN_K     256
#define PQN_ALPHA 10.0f
// w-transpose LDS row stride in halves: 264 -> 132 dwords == 4 mod 64 (conflict-free)
#define WSTR      264

// ---------------------------------------------------------------------------
// Prep: f32 codebook c[1024][256] -> f16 WMMA B-fragments in workspace.
//   B1 frag id = m*64 + s*16 + t : value = c[(m*128 + s*32+kk)*256 + t*16+n]
//   B2 frag id = 512 + m*64 + r*8 + u : value = c[(m*128 + u*16+n)*256 + r*32+kk]
// (B layout: lane = column n (lane&15), half = lane>>4, elem i -> kk = (i>>3)*16+half*8+(i&7))
// ---------------------------------------------------------------------------
__global__ __launch_bounds__(256) void pqn_prep(const float* __restrict__ c,
                                                _Float16* __restrict__ bfrag) {
  const int lane = threadIdx.x & 31;
  const int wave = threadIdx.x >> 5;
  const int frag = blockIdx.x * 8 + wave;      // 0..1023
  const int n16  = lane & 15;
  const int half = lane >> 4;

  v16h hv;
  if (frag < 512) {
    const int m = frag >> 6, s = (frag >> 4) & 3, t = frag & 15;
    const int n = t * 16 + n16;
#pragma unroll
    for (int i = 0; i < 16; ++i) {
      const int l = s * 32 + (i >> 3) * 16 + half * 8 + (i & 7);
      hv[i] = (_Float16)c[(size_t)(m * PQN_L + l) * PQN_K + n];
    }
  } else {
    const int g = frag - 512;
    const int m = g >> 6, r = (g >> 3) & 7, u = g & 7;
    const int lcol = u * 16 + n16;
#pragma unroll
    for (int i = 0; i < 16; ++i) {
      const int k = r * 32 + (i >> 3) * 16 + half * 8 + (i & 7);
      hv[i] = (_Float16)c[(size_t)(m * PQN_L + lcol) * PQN_K + k];
    }
  }
  *(v16h*)(bfrag + (size_t)frag * 512 + lane * 16) = hv;
}

// ---------------------------------------------------------------------------
// TDM: async-copy 64 KB contiguous (8192 x 8B elements, 1-row tile) global->LDS.
// ---------------------------------------------------------------------------
__device__ __forceinline__ void tdm_copy_64kb(const _Float16* gsrc, _Float16* ldst) {
  const unsigned ldsOff = (unsigned)(uintptr_t)ldst;        // low 32 bits = LDS offset
  const unsigned long long ga = (unsigned long long)(uintptr_t)gsrc;
  u32x4 g0;
  g0[0] = 1u;                                   // count=1, user descriptor
  g0[1] = ldsOff;                               // lds_addr (bytes)
  g0[2] = (unsigned)(ga & 0xffffffffu);         // global_addr[31:0]
  g0[3] = (unsigned)((ga >> 32) & 0x01ffffffu)  // global_addr[56:32]
          | (2u << 30);                         // type = 2 ("image")
  i32x8 g1;
  g1[0] = (int)(3u << 16);                      // data_size = 3 (8 bytes)
  g1[1] = (int)((8192u & 0xffffu) << 16);       // tensor_dim0 = 8192 (lo16)
  g1[2] = (int)(1u << 16);                      // dim0 hi16=0 | tensor_dim1 = 1
  g1[3] = (int)(8192u << 16);                   // dim1 hi16=0 | tile_dim0 = 8192
  g1[4] = 1;                                    // tile_dim1 = 1, tile_dim2 = 0
  g1[5] = 8192;                                 // tensor_dim0_stride lo32
  g1[6] = 0;
  g1[7] = 0;
  const i32x4 z4 = {0, 0, 0, 0};
  const i32x8 z8 = {0, 0, 0, 0, 0, 0, 0, 0};
  __builtin_amdgcn_tensor_load_to_lds(g0, g1, z4, z4, z8, 0);
}

// ---------------------------------------------------------------------------
// Main: block = 8 waves, all subspace m = blockIdx.y; wave w handles row-tile
// blockIdx.x*8 + w (16 rows). B fragments staged per block into LDS via TDM
// (B1 then B2, same 64 KB buffer) and shared by all 8 waves.
// ---------------------------------------------------------------------------
__global__ __launch_bounds__(256) void pqn_main(const float* __restrict__ x,
                                                const _Float16* __restrict__ bfrag,
                                                float* __restrict__ out) {
  __shared__ __attribute__((aligned(16))) _Float16 frag[32768];        // 64 KB B stage
  __shared__ __attribute__((aligned(16))) _Float16 wbuf[8 * 16 * WSTR]; // 66 KB w

  const int lane  = threadIdx.x & 31;
  const int wave  = threadIdx.x >> 5;
  const int m     = blockIdx.y;
  const int rtile = blockIdx.x * 8 + wave;
  const int n16   = lane & 15;
  const int half  = lane >> 4;

  // ---- kick off async B1 staging (wave 0; TDM ignores EXEC, issues once)
  if (threadIdx.x == 0) {
    tdm_copy_64kb(bfrag + (size_t)m * 64 * 512, frag);
  }

  // ---- meanwhile: load this lane's A-row of x + sum-of-squares
  const float* xrow = x + (size_t)(rtile * 16 + n16) * PQN_D + m * PQN_L;
  float xa[64];
  float ss = 0.f;
#pragma unroll
  for (int s = 0; s < 4; ++s) {
#pragma unroll
    for (int c2 = 0; c2 < 2; ++c2) {
      const float4 v0 = *(const float4*)(xrow + s * 32 + c2 * 16 + half * 8);
      const float4 v1 = *(const float4*)(xrow + s * 32 + c2 * 16 + half * 8 + 4);
      const int b = s * 16 + c2 * 8;
      xa[b + 0] = v0.x; xa[b + 1] = v0.y; xa[b + 2] = v0.z; xa[b + 3] = v0.w;
      xa[b + 4] = v1.x; xa[b + 5] = v1.y; xa[b + 6] = v1.z; xa[b + 7] = v1.w;
      ss += v0.x * v0.x + v0.y * v0.y + v0.z * v0.z + v0.w * v0.w;
      ss += v1.x * v1.x + v1.y * v1.y + v1.z * v1.z + v1.w * v1.w;
    }
  }
  ss += __shfl_xor(ss, 16, 32);
  const float scale = (2.0f * PQN_ALPHA) / fmaxf(sqrtf(ss), 1e-12f);

  // ---- A fragments for GEMM1
  v16h a1[4];
#pragma unroll
  for (int s = 0; s < 4; ++s)
#pragma unroll
    for (int i = 0; i < 16; ++i)
      a1[s][i] = (_Float16)(xa[s * 16 + i] * scale);

  if (threadIdx.x == 0) __builtin_amdgcn_s_wait_tensorcnt(0);
  __syncthreads();                               // B1 visible to all waves

  // ---- GEMM1: logits[16 rows][256 codes]; two independent acc chains per
  //      t-pair, B fragments prefetched one iteration ahead.
  const _Float16* fb = frag + lane * 16;
  v8f logit[16];
  {
    v16h b0 = *(const v16h*)(fb + (0 * 16 + 0) * 512);   // (s=0, t=0)
    v16h b1v = *(const v16h*)(fb + (0 * 16 + 1) * 512);  // (s=0, t=1)
#pragma unroll
    for (int tp = 0; tp < 8; ++tp) {
      v8f acc0 = {}, acc1 = {};
#pragma unroll
      for (int s = 0; s < 4; ++s) {
        v16h n0 = b0, n1 = b1v;
        const int ns = (s + 1) & 3;
        const int ntp = (s == 3) ? tp + 1 : tp;
        if (ntp < 8) {                           // compile-time after unroll
          n0 = *(const v16h*)(fb + (ns * 16 + ntp * 2 + 0) * 512);
          n1 = *(const v16h*)(fb + (ns * 16 + ntp * 2 + 1) * 512);
        }
        acc0 = __builtin_amdgcn_wmma_f32_16x16x32_f16(false, a1[s], false, b0,
                                                      (short)0, acc0, false, false);
        acc1 = __builtin_amdgcn_wmma_f32_16x16x32_f16(false, a1[s], false, b1v,
                                                      (short)0, acc1, false, false);
        b0 = n0; b1v = n1;
      }
      logit[tp * 2 + 0] = acc0;
      logit[tp * 2 + 1] = acc1;
    }
  }

  // ---- softmax over K=256 (row r lives in VGPR r across 16 lanes of a half)
  float mx[8], sm[8], inv[8];
#pragma unroll
  for (int r = 0; r < 8; ++r) mx[r] = -3.4e38f;
#pragma unroll
  for (int t = 0; t < 16; ++t)
#pragma unroll
    for (int r = 0; r < 8; ++r) mx[r] = fmaxf(mx[r], logit[t][r]);
#pragma unroll
  for (int r = 0; r < 8; ++r) {
    mx[r] = fmaxf(mx[r], __shfl_xor(mx[r], 1, 32));
    mx[r] = fmaxf(mx[r], __shfl_xor(mx[r], 2, 32));
    mx[r] = fmaxf(mx[r], __shfl_xor(mx[r], 4, 32));
    mx[r] = fmaxf(mx[r], __shfl_xor(mx[r], 8, 32));
    sm[r] = 0.f;
  }
#pragma unroll
  for (int t = 0; t < 16; ++t)
#pragma unroll
    for (int r = 0; r < 8; ++r) {
      const float e = __expf(logit[t][r] - mx[r]);
      logit[t][r] = e;
      sm[r] += e;
    }
#pragma unroll
  for (int r = 0; r < 8; ++r) {
    sm[r] += __shfl_xor(sm[r], 1, 32);
    sm[r] += __shfl_xor(sm[r], 2, 32);
    sm[r] += __shfl_xor(sm[r], 4, 32);
    sm[r] += __shfl_xor(sm[r], 8, 32);
    inv[r] = 1.0f / sm[r];
  }

  // ---- transpose w through LDS (C-layout regs -> [row][code] f16)
  _Float16* wl = wbuf + wave * 16 * WSTR;
#pragma unroll
  for (int t = 0; t < 16; ++t)
#pragma unroll
    for (int r = 0; r < 8; ++r)
      wl[(half * 8 + r) * WSTR + t * 16 + n16] = (_Float16)(logit[t][r] * inv[r]);

  __syncthreads();                               // all waves done reading B1
  if (threadIdx.x == 0) {
    tdm_copy_64kb(bfrag + (size_t)(512 + m * 64) * 512, frag);  // stage B2
    __builtin_amdgcn_s_wait_tensorcnt(0);
  }
  __syncthreads();                               // B2 visible to all waves

  // ---- A fragments for GEMM2 (w, contraction over codes) from LDS
  v16h a2[8];
#pragma unroll
  for (int r2 = 0; r2 < 8; ++r2) {
    const _Float16* p = wl + n16 * WSTR + r2 * 32 + half * 8;
    const v8h lo = *(const v8h*)(p);             // kk = half*8 + 0..7
    const v8h hi = *(const v8h*)(p + 16);        // kk = 16 + half*8 + 0..7
    a2[r2] = __builtin_shufflevector(lo, hi, 0, 1, 2, 3, 4, 5, 6, 7,
                                     8, 9, 10, 11, 12, 13, 14, 15);
  }

  // ---- GEMM2 (u-pairs, prefetched) + store
  float* orow = out + (size_t)(rtile * 16 + half * 8) * PQN_D + m * PQN_L;
  {
    v16h c0 = *(const v16h*)(fb + (0 * 8 + 0) * 512);    // (r2=0, u=0)
    v16h c1 = *(const v16h*)(fb + (0 * 8 + 1) * 512);    // (r2=0, u=1)
#pragma unroll
    for (int up = 0; up < 4; ++up) {
      v8f acc0 = {}, acc1 = {};
#pragma unroll
      for (int r2 = 0; r2 < 8; ++r2) {
        v16h n0 = c0, n1 = c1;
        const int nr = (r2 + 1) & 7;
        const int nup = (r2 == 7) ? up + 1 : up;
        if (nup < 4) {                           // compile-time after unroll
          n0 = *(const v16h*)(fb + (nr * 8 + nup * 2 + 0) * 512);
          n1 = *(const v16h*)(fb + (nr * 8 + nup * 2 + 1) * 512);
        }
        acc0 = __builtin_amdgcn_wmma_f32_16x16x32_f16(false, a2[r2], false, c0,
                                                      (short)0, acc0, false, false);
        acc1 = __builtin_amdgcn_wmma_f32_16x16x32_f16(false, a2[r2], false, c1,
                                                      (short)0, acc1, false, false);
        c0 = n0; c1 = n1;
      }
#pragma unroll
      for (int r = 0; r < 8; ++r) {
        orow[(size_t)r * PQN_D + (up * 2 + 0) * 16 + n16] = acc0[r];
        orow[(size_t)r * PQN_D + (up * 2 + 1) * 16 + n16] = acc1[r];
      }
    }
  }
}

extern "C" void kernel_launch(void* const* d_in, const int* in_sizes, int n_in,
                              void* d_out, int out_size, void* d_ws, size_t ws_size,
                              hipStream_t stream) {
  const float* x = (const float*)d_in[0];   // [32768, 1024] f32
  const float* c = (const float*)d_in[1];   // [1024, 256]  f32
  float* out = (float*)d_out;               // [32768, 1024] f32
  _Float16* bfrag = (_Float16*)d_ws;        // 1 MB f16 fragments

  pqn_prep<<<128, 256, 0, stream>>>(c, bfrag);
  pqn_main<<<dim3(256, 8), 256, 0, stream>>>(x, bfrag, out);
}
